// LearnNMSModule_64707977282216
// MI455X (gfx1250) — matrix operand order repaired and checked by reading, compiled, not run on hardware
//
#include <hip/hip_runtime.h>
#include <hip/hip_bf16.h>
#include <math.h>

// ---------------- problem constants ----------------
#define B_        2
#define NB_       512
#define C_        80
#define FEAT_     1024
#define FIRST_N_  128
#define NMS_FC_   128
#define GROUPS_   16
#define DQ_       64          // ATT_D[0]/GROUPS
#define SCALE_CLAMP_ 4.135166556742356f   // log(1000/16)

typedef __attribute__((ext_vector_type(16))) _Float16 v16h;
typedef __attribute__((ext_vector_type(8)))  float    v8f;
typedef __attribute__((ext_vector_type(4)))  int      v4i;

// 100 / 1000^(j/8), j=0..7  (positional-embedding inverse frequencies * 100)
__device__ __constant__ float c_invdim[8] = {
    100.0f, 42.169650f, 17.782794f, 7.4989421f,
    3.1622777f, 1.3335214f, 0.56234133f, 0.23713737f};

// ---------------- gfx1250 async global->LDS copy (ASYNCcnt path) ----------------
#if defined(__has_builtin)
#if __has_builtin(__builtin_amdgcn_global_load_async_to_lds_b128)
#define ASYNC_COPY 1
#endif
#endif

__device__ inline void async_wait0() {
#if defined(__has_builtin) && __has_builtin(__builtin_amdgcn_s_wait_asynccnt)
  __builtin_amdgcn_s_wait_asynccnt(0);
#else
  asm volatile("s_wait_asynccnt 0x0" ::: "memory");
#endif
}

#ifdef ASYNC_COPY
// one 16-byte async copy per lane: LDS[l] = MEM[g]; tracked by ASYNCcnt.
// ROCm 7.2 signature (from diagnostics): (v4i AS1*, v4i AS3*, imm offset, imm cpol)
__device__ inline void async_copy_b128(const void* g, void* l) {
  __builtin_amdgcn_global_load_async_to_lds_b128(
      (__attribute__((address_space(1))) v4i*)(g),
      (__attribute__((address_space(3))) v4i*)(l), 0, 0);
}
#endif

// ============================================================
// WMMA fragment loaders per CDNA5 ISA 7.12.2 (wave32, 16x16x32 f16)
//   A (16x32): lanes 0-15 => M=0..15 (lanes 16-31 same M, +8 in K);
//              VGPR v<4: K=2v+s+8*half ; v>=4: K=16+2(v-4)+s+8*half
//   B (32x16): lane => N = lane&15; slot e => K same as A (e<8: e+8*half; else 16+8*half+e-8)
//   C/D f32  : VGPR r => M = r + 8*(lane>>4); N = lane&15
// ============================================================
template <typename T>
__device__ inline v16h frag_a(const T* A, int lda, int row0, int k0) {
  const int lane = threadIdx.x & 31;
  const int hlf  = lane >> 4;
  const int m    = lane & 15;
  v16h a;
#pragma unroll
  for (int e = 0; e < 16; ++e) {
    const int v = e >> 1, s = e & 1;
    const int k = (v < 4) ? (2 * v + s + 8 * hlf) : (16 + 2 * (v - 4) + s + 8 * hlf);
    a[e] = (_Float16)A[(row0 + m) * lda + k0 + k];
  }
  return a;
}

template <typename T>
__device__ inline v16h frag_b(const T* Bm, int ldb, int k0, int col0) {
  const int lane = threadIdx.x & 31;
  const int hlf = lane >> 4, n = lane & 15;
  v16h b;
#pragma unroll
  for (int e = 0; e < 16; ++e) {
    const int k = k0 + ((e < 8) ? (e + 8 * hlf) : (16 + 8 * hlf + e - 8));
    b[e] = (_Float16)Bm[k * ldb + col0 + n];
  }
  return b;
}

// B = S^T where S is (N x K) row-major  (used for q @ k^T)
template <typename T>
__device__ inline v16h frag_bT(const T* S, int ldS, int k0, int col0) {
  const int lane = threadIdx.x & 31;
  const int hlf = lane >> 4, n = lane & 15;
  v16h b;
#pragma unroll
  for (int e = 0; e < 16; ++e) {
    const int k = k0 + ((e < 8) ? (e + 8 * hlf) : (16 + 8 * hlf + e - 8));
    b[e] = (_Float16)S[(col0 + n) * ldS + k];
  }
  return b;
}

// ============================================================
// K0: box refinement (apply_deltas)
// ============================================================
__global__ void refine_kernel(const float* __restrict__ deltas,
                              const float* __restrict__ boxes,
                              float* __restrict__ refined) {
  int gid = blockIdx.x * blockDim.x + threadIdx.x;
  if (gid >= B_ * NB_ * C_) return;
  int row = gid / C_, c = gid % C_;
  float x1 = boxes[row * 4 + 0], y1 = boxes[row * 4 + 1];
  float x2 = boxes[row * 4 + 2], y2 = boxes[row * 4 + 3];
  float w = x2 - x1, h = y2 - y1;
  float cx = x1 + 0.5f * w, cy = y1 + 0.5f * h;
  const float* d = &deltas[row * (C_ * 4) + c * 4];
  float dx = d[0] * 0.1f, dy = d[1] * 0.1f;
  float dw = fminf(d[2] * 0.2f, SCALE_CLAMP_);
  float dh = fminf(d[3] * 0.2f, SCALE_CLAMP_);
  float pcx = dx * w + cx, pcy = dy * h + cy;
  float pw = __expf(dw) * w, ph = __expf(dh) * h;
  float* o = &refined[(size_t)gid * 4];
  o[0] = pcx - 0.5f * pw; o[1] = pcy - 0.5f * ph;
  o[2] = pcx + 0.5f * pw; o[3] = pcy + 0.5f * ph;
}

// ============================================================
// K1: fused per-(b,c) softmax + descending bitonic top-128 + box gather
//     (softmax is monotone => sort raw logits, normalize the survivors)
// ============================================================
__global__ void __launch_bounds__(256)
softmax_sort_kernel(const float* __restrict__ scores,
                    const float* __restrict__ refined,
                    float* __restrict__ sorted_score,
                    int*   __restrict__ rank_idx,
                    float* __restrict__ sorted_boxes) {
  __shared__ float key[NB_];
  __shared__ int   val[NB_];
  __shared__ float red[256];
  const int b = blockIdx.x / C_, c = blockIdx.x % C_;
  const int tid = threadIdx.x;
  for (int n = tid; n < NB_; n += 256) {
    key[n] = scores[(size_t)(b * NB_ + n) * (C_ + 1) + c];
    val[n] = n;
  }
  __syncthreads();
  red[tid] = fmaxf(key[tid], key[tid + 256]);
  __syncthreads();
  for (int s = 128; s > 0; s >>= 1) {
    if (tid < s) red[tid] = fmaxf(red[tid], red[tid + s]);
    __syncthreads();
  }
  const float mx = red[0];
  __syncthreads();
  red[tid] = __expf(key[tid] - mx) + __expf(key[tid + 256] - mx);
  __syncthreads();
  for (int s = 128; s > 0; s >>= 1) {
    if (tid < s) red[tid] += red[tid + s];
    __syncthreads();
  }
  const float denom = red[0];
  // bitonic sort, descending
  for (int kk = 2; kk <= NB_; kk <<= 1) {
    for (int j = kk >> 1; j > 0; j >>= 1) {
      __syncthreads();
      for (int t = tid; t < NB_; t += 256) {
        int ixj = t ^ j;
        if (ixj > t) {
          bool desc = ((t & kk) == 0);
          float a = key[t], b2 = key[ixj];
          if (desc ? (a < b2) : (a > b2)) {
            key[t] = b2; key[ixj] = a;
            int tv = val[t]; val[t] = val[ixj]; val[ixj] = tv;
          }
        }
      }
    }
  }
  __syncthreads();
  if (tid < FIRST_N_) {
    float p = __expf(key[tid] - mx) / denom;
    int src = val[tid];
    sorted_score[(b * FIRST_N_ + tid) * C_ + c] = p;
    rank_idx[(b * FIRST_N_ + tid) * C_ + c]     = src;
    const float* rb = &refined[((size_t)(b * NB_ + src) * C_ + c) * 4];
    float* ob = &sorted_boxes[((size_t)(b * FIRST_N_ + tid) * C_ + c) * 4];
    ob[0] = rb[0]; ob[1] = rb[1]; ob[2] = rb[2]; ob[3] = rb[3];
  }
}

// ============================================================
// K2: f32 -> f16 convert
// ============================================================
__global__ void f32_to_f16(const float* __restrict__ x, _Float16* __restrict__ y, int n) {
  int i = blockIdx.x * blockDim.x + threadIdx.x;
  if (i < n) y[i] = (_Float16)x[i];
}

// K3: sinusoidal rank embedding -> f16 (128 x 1024)
__global__ void rank_embed_kernel(_Float16* __restrict__ re) {
  int gid = blockIdx.x * blockDim.x + threadIdx.x;
  if (gid >= FIRST_N_ * FEAT_) return;
  int i = gid / FEAT_, j = gid % FEAT_;
  int jj = (j < 512) ? j : (j - 512);
  float ang = (float)i * __powf(1000.0f, -(float)jj / 512.0f);
  re[gid] = (_Float16)((j < 512) ? __sinf(ang) : __cosf(ang));
}

// ============================================================
// K4: generic WMMA GEMM  C(MxN)=A(MxK)@B(KxN)+bias ; f16 in, f32 out
//     block = 128 threads (4 waves) computing a 32x32 tile, kc=32.
//     Tiles staged to LDS via GLOBAL_LOAD_ASYNC_TO_LDS_B128 (ASYNCcnt)
//     when available; LDS row stride 40 halves = 80B keeps every
//     8-half chunk 16B-aligned for the B128 async writes.
// ============================================================
__global__ void __launch_bounds__(128)
gemm_f16_wmma(const _Float16* __restrict__ A, const _Float16* __restrict__ Bm,
              const float* __restrict__ bias, float* __restrict__ Cm,
              int M, int N, int K) {
  __shared__ _Float16 As[32 * 40];
  __shared__ _Float16 Bs[32 * 40];
  const int tid  = threadIdx.x;
  const int wave = tid >> 5;
  const int lane = tid & 31;
  const int hlf  = lane >> 4;
  const int m16  = lane & 15;
  const int bm = blockIdx.x * 32;
  const int bn = blockIdx.y * 32;
  const int wm = (wave >> 1) * 16;
  const int wn = (wave & 1) * 16;
  const int er = (tid * 8) >> 5;     // this thread's tile row
  const int ec = (tid * 8) & 31;     // this thread's tile col (0,8,16,24)
  v8f acc = {};
  for (int k0 = 0; k0 < K; k0 += 32) {
    __syncthreads();
#ifdef ASYNC_COPY
    async_copy_b128(&A[(size_t)(bm + er) * K + k0 + ec], &As[er * 40 + ec]);
    async_copy_b128(&Bm[(size_t)(k0 + er) * N + bn + ec], &Bs[er * 40 + ec]);
    async_wait0();
#else
#pragma unroll
    for (int i = 0; i < 8; ++i) {
      int e = tid * 8 + i;
      int r = e >> 5, col = e & 31;
      As[r * 40 + col] = A[(size_t)(bm + r) * K + k0 + col];
      Bs[r * 40 + col] = Bm[(size_t)(k0 + r) * N + bn + col];
    }
#endif
    __syncthreads();
    if (k0 + 64 < K) {                       // global_prefetch_b8 of next tiles
      __builtin_prefetch(&A[(size_t)bm * K + k0 + 64], 0, 1);
      __builtin_prefetch(&Bm[(size_t)(k0 + 64) * N + bn], 0, 1);
    }
    v16h a = frag_a(As, 40, wm, 0);
    v16h b = frag_b(Bs, 40, 0, wn);
    acc = __builtin_amdgcn_wmma_f32_16x16x32_f16(false, a, false, b, (short)0,
                                                 acc, false, false);
  }
  const float bv = bias ? bias[bn + wn + m16] : 0.0f;
#pragma unroll
  for (int r = 0; r < 8; ++r)
    Cm[(size_t)(bm + wm + r + 8 * hlf) * N + bn + wn + m16] = acc[r] + bv;
  (void)M;
}

// ============================================================
// K5: gather sorted roi embedding + rank embedding -> emb (f32 and f16)
//     layout (b, c, n, fc)
// ============================================================
__global__ void gather_emb_kernel(const float* __restrict__ roi_emb,
                                  const float* __restrict__ rank_feat,
                                  const int*   __restrict__ rank_idx,
                                  float* __restrict__ emb_f32,
                                  _Float16* __restrict__ emb_h) {
  int gid = blockIdx.x * blockDim.x + threadIdx.x;
  if (gid >= B_ * C_ * FIRST_N_ * NMS_FC_) return;
  int fc = gid & 127;
  int n  = (gid >> 7) & 127;
  int rest = gid >> 14;
  int c = rest % C_, b = rest / C_;
  int src = rank_idx[(b * FIRST_N_ + n) * C_ + c];
  float v = roi_emb[(size_t)(b * NB_ + src) * NMS_FC_ + fc] + rank_feat[n * NMS_FC_ + fc];
  emb_f32[gid] = v;
  emb_h[gid]   = (_Float16)v;
}

// ============================================================
// K6: fused relation-attention per (b,c).  One workgroup (8 waves),
//     ~200KB dynamic LDS (CDNA5 WGP has 320KB) keeps f, q, k, aff, ov
//     resident: avoids ~300MB of HBM intermediates.
// ============================================================
__global__ void __launch_bounds__(256)
attention_kernel(const _Float16* __restrict__ emb_h,
                 const float* __restrict__ sorted_boxes,
                 const _Float16* __restrict__ qw_h, const float* __restrict__ qb,
                 const _Float16* __restrict__ kw_h, const float* __restrict__ kb,
                 const float* __restrict__ posw, const float* __restrict__ posb,
                 const float* __restrict__ outw,
                 float* __restrict__ att) {
  extern __shared__ char smem[];
  _Float16* fh   = (_Float16*)(smem);            // 128x128 f16   32KB
  _Float16* qh   = (_Float16*)(smem + 32768);    // 128x64  f16   16KB
  _Float16* kh   = (_Float16*)(smem + 49152);    // 128x64  f16   16KB
  float*    sm   = (float*)(smem + 65536);       // 128x128 f32   64KB
  float*    ov   = (float*)(smem + 131072);      // 128x128 f32   64KB
  float*    geom = (float*)(smem + 196608);      // 128x6   f32    3KB
  float*    pwc  = (float*)(smem + 199680);      // 64      f32  256B

  const int bc = blockIdx.x;
  const int b = bc / C_, c = bc % C_;
  const int tid  = threadIdx.x;
  const int wave = tid >> 5;
  const int lane = tid & 31;
  const int hlf  = lane >> 4;
  const int m16  = lane & 15;

  const _Float16* fsrc = &emb_h[(size_t)bc * FIRST_N_ * NMS_FC_];
#ifdef ASYNC_COPY
  for (int i = tid * 8; i < FIRST_N_ * NMS_FC_; i += 256 * 8)
    async_copy_b128(&fsrc[i], &fh[i]);           // 32KB tile via ASYNCcnt DMA
  async_wait0();
#else
  for (int i = tid; i < FIRST_N_ * NMS_FC_; i += 256) fh[i] = fsrc[i];
#endif
  for (int i = tid; i < FIRST_N_; i += 256) {
    const float* bx = &sorted_boxes[((size_t)(b * FIRST_N_ + i) * C_ + c) * 4];
    float w = bx[2] - bx[0] + 1.0f, h = bx[3] - bx[1] + 1.0f;
    geom[i * 6 + 0] = w;                     geom[i * 6 + 1] = h;
    geom[i * 6 + 2] = 0.5f * (bx[0] + bx[2]); geom[i * 6 + 3] = 0.5f * (bx[1] + bx[3]);
    geom[i * 6 + 4] = __logf(w);              geom[i * 6 + 5] = __logf(h);
  }
  __syncthreads();

  for (int g = 0; g < GROUPS_; ++g) {
    if (tid < 64) pwc[tid] = posw[tid * GROUPS_ + g];   // pos_w column g
    // ---- q/k projections: (128x128) @ (128x64) with WMMA ----
    for (int tt = wave; tt < 32; tt += 8) {
      int tm = tt & 7, tn = tt >> 3;
      v8f aq = {}, ak = {};
      for (int k0 = 0; k0 < NMS_FC_; k0 += 32) {
        v16h af = frag_a(fh, NMS_FC_, tm * 16, k0);
        v16h bq = frag_b(qw_h, GROUPS_ * DQ_, k0, g * DQ_ + tn * 16);
        aq = __builtin_amdgcn_wmma_f32_16x16x32_f16(false, af, false, bq, (short)0, aq, false, false);
        v16h bk = frag_b(kw_h, GROUPS_ * DQ_, k0, g * DQ_ + tn * 16);
        ak = __builtin_amdgcn_wmma_f32_16x16x32_f16(false, af, false, bk, (short)0, ak, false, false);
      }
      int col = tn * 16 + m16;
      float bqv = qb[g * DQ_ + col], bkv = kb[g * DQ_ + col];
#pragma unroll
      for (int r = 0; r < 8; ++r) {
        int row = tm * 16 + r + 8 * hlf;
        qh[row * DQ_ + col] = (_Float16)(aq[r] + bqv);
        kh[row * DQ_ + col] = (_Float16)(ak[r] + bkv);
      }
    }
    __syncthreads();
    // ---- aff = q @ k^T / sqrt(dk) ----
    for (int tt = wave; tt < 64; tt += 8) {
      int tm = tt & 7, tn = tt >> 3;
      v8f acc = {};
      for (int k0 = 0; k0 < DQ_; k0 += 32) {
        v16h a  = frag_a(qh, DQ_, tm * 16, k0);
        v16h bt = frag_bT(kh, DQ_, k0, tn * 16);
        acc = __builtin_amdgcn_wmma_f32_16x16x32_f16(false, a, false, bt, (short)0, acc, false, false);
      }
#pragma unroll
      for (int r = 0; r < 8; ++r)
        sm[(tm * 16 + r + 8 * hlf) * 128 + tn * 16 + m16] = acc[r] * 0.125f;
    }
    __syncthreads();
    // ---- + log(max(relu(pe @ pos_w + pos_b), 1e-6)) (computed in-place) ----
    for (int idx = tid; idx < FIRST_N_ * FIRST_N_; idx += 256) {
      int n = idx >> 7, m2 = idx & 127;
      float pmv[4];
      pmv[0] = 100.f * __logf(fmaxf(fabsf((geom[n*6+2] - geom[m2*6+2]) / geom[n*6+0]), 1e-3f));
      pmv[1] = 100.f * __logf(fmaxf(fabsf((geom[n*6+3] - geom[m2*6+3]) / geom[n*6+1]), 1e-3f));
      pmv[2] = 100.f * (geom[n*6+4] - geom[m2*6+4]);
      pmv[3] = 100.f * (geom[n*6+5] - geom[m2*6+5]);
      float accg = posb[g];
#pragma unroll
      for (int comp = 0; comp < 4; ++comp) {
#pragma unroll
        for (int j = 0; j < 8; ++j) {
          float ang = pmv[comp] * c_invdim[j];
          accg += __sinf(ang) * pwc[comp * 16 + j] + __cosf(ang) * pwc[comp * 16 + 8 + j];
        }
      }
      sm[idx] += __logf(fmaxf(fmaxf(accg, 0.0f), 1e-6f));
    }
    __syncthreads();
    // ---- row softmax ----
    if (tid < FIRST_N_) {
      float* row = &sm[tid * 128];
      float mx = -3.0e38f;
      for (int m2 = 0; m2 < 128; ++m2) mx = fmaxf(mx, row[m2]);
      float s = 0.f;
      for (int m2 = 0; m2 < 128; ++m2) { float e = __expf(row[m2] - mx); row[m2] = e; s += e; }
      float inv = 1.0f / s;
      for (int m2 = 0; m2 < 128; ++m2) row[m2] *= inv;
    }
    __syncthreads();
    // ---- ov = sm @ f ----
    for (int tt = wave; tt < 64; tt += 8) {
      int tm = tt & 7, tn = tt >> 3;
      v8f acc = {};
      for (int k0 = 0; k0 < NMS_FC_; k0 += 32) {
        v16h a = frag_a(sm, 128, tm * 16, k0);   // f32 LDS -> f16 frag
        v16h b2 = frag_b(fh, NMS_FC_, k0, tn * 16);
        acc = __builtin_amdgcn_wmma_f32_16x16x32_f16(false, a, false, b2, (short)0, acc, false, false);
      }
#pragma unroll
      for (int r = 0; r < 8; ++r)
        ov[(tm * 16 + r + 8 * hlf) * 128 + tn * 16 + m16] = acc[r];
    }
    __syncthreads();
    // ---- grouped output projection: att[.., g*8+e] = ov @ out_w[g] ----
    for (int idx = tid; idx < FIRST_N_ * 8; idx += 256) {
      int n = idx >> 3, e = idx & 7;
      const float* wrow = &outw[(size_t)(g * 128) * 8 + e];
      const float* orow = &ov[n * 128];
      float acc = 0.f;
      for (int d = 0; d < 128; ++d) acc += orow[d] * wrow[d * 8];
      att[((size_t)bc * FIRST_N_ + n) * NMS_FC_ + g * 8 + e] = acc;
    }
    __syncthreads();
  }
}

// ============================================================
// K7: all_feat = relu(emb + att + out_b); logits; sigmoid * sorted_score
// ============================================================
__global__ void final_kernel(const float* __restrict__ emb_f32,
                             const float* __restrict__ att,
                             const float* __restrict__ outb,
                             const float* __restrict__ logit_w,
                             const float* __restrict__ logit_b,
                             const float* __restrict__ sorted_score,
                             float* __restrict__ out) {
  int gid = blockIdx.x * blockDim.x + threadIdx.x;
  if (gid >= B_ * C_ * FIRST_N_) return;
  int n = gid % FIRST_N_;
  int bc = gid / FIRST_N_;
  int c = bc % C_, b = bc / C_;
  const float* e = &emb_f32[(size_t)gid * NMS_FC_];
  const float* a = &att[(size_t)gid * NMS_FC_];
  float l0 = logit_b[0], l1 = logit_b[1], l2 = logit_b[2];
  for (int fc = 0; fc < NMS_FC_; ++fc) {
    float v = fmaxf(e[fc] + a[fc] + outb[fc], 0.f);
    l0 += v * logit_w[fc * 3 + 0];
    l1 += v * logit_w[fc * 3 + 1];
    l2 += v * logit_w[fc * 3 + 2];
  }
  float ss = sorted_score[(b * FIRST_N_ + n) * C_ + c];
  float* o = &out[((size_t)(b * FIRST_N_ + n) * C_ + c) * 3];
  o[0] = ss / (1.f + __expf(-l0));
  o[1] = ss / (1.f + __expf(-l1));
  o[2] = ss / (1.f + __expf(-l2));
}

// ============================================================
// launcher
// ============================================================
extern "C" void kernel_launch(void* const* d_in, const int* in_sizes, int n_in,
                              void* d_out, int out_size, void* d_ws, size_t ws_size,
                              hipStream_t stream) {
  (void)in_sizes; (void)n_in; (void)out_size; (void)ws_size;
  const float* roi_feat = (const float*)d_in[0];
  const float* scores   = (const float*)d_in[1];
  const float* deltas   = (const float*)d_in[2];
  const float* pboxes   = (const float*)d_in[3];
  const float* roi_w    = (const float*)d_in[4];
  const float* roi_b    = (const float*)d_in[5];
  const float* rank_w   = (const float*)d_in[6];
  const float* rank_b   = (const float*)d_in[7];
  const float* logit_w  = (const float*)d_in[8];
  const float* logit_b  = (const float*)d_in[9];
  const float* pos_w    = (const float*)d_in[10];
  const float* pos_b    = (const float*)d_in[11];
  const float* q_w      = (const float*)d_in[12];
  const float* q_b      = (const float*)d_in[13];
  const float* k_w      = (const float*)d_in[14];
  const float* k_b      = (const float*)d_in[15];
  const float* out_w    = (const float*)d_in[16];
  const float* out_b    = (const float*)d_in[17];
  float* out = (float*)d_out;

  char* p = (char*)d_ws;
  auto alloc = [&](size_t bytes) -> void* {
    void* r = (void*)p;
    p += (bytes + 255) & ~(size_t)255;
    return r;
  };
  float*    refined      = (float*)alloc((size_t)B_ * NB_ * C_ * 4 * sizeof(float));
  float*    sorted_score = (float*)alloc((size_t)B_ * FIRST_N_ * C_ * sizeof(float));
  int*      rank_idx     = (int*)  alloc((size_t)B_ * FIRST_N_ * C_ * sizeof(int));
  float*    sorted_boxes = (float*)alloc((size_t)B_ * FIRST_N_ * C_ * 4 * sizeof(float));
  _Float16* roi_feat_h   = (_Float16*)alloc((size_t)B_ * NB_ * FEAT_ * sizeof(_Float16));
  _Float16* roi_w_h      = (_Float16*)alloc((size_t)FEAT_ * NMS_FC_ * sizeof(_Float16));
  _Float16* rank_w_h     = (_Float16*)alloc((size_t)FEAT_ * NMS_FC_ * sizeof(_Float16));
  _Float16* qw_h         = (_Float16*)alloc((size_t)NMS_FC_ * 1024 * sizeof(_Float16));
  _Float16* kw_h         = (_Float16*)alloc((size_t)NMS_FC_ * 1024 * sizeof(_Float16));
  _Float16* re_h         = (_Float16*)alloc((size_t)FIRST_N_ * FEAT_ * sizeof(_Float16));
  float*    roi_emb      = (float*)alloc((size_t)B_ * NB_ * NMS_FC_ * sizeof(float));
  float*    rank_feat    = (float*)alloc((size_t)FIRST_N_ * NMS_FC_ * sizeof(float));
  float*    emb_f32      = (float*)alloc((size_t)B_ * C_ * FIRST_N_ * NMS_FC_ * sizeof(float));
  _Float16* emb_h        = (_Float16*)alloc((size_t)B_ * C_ * FIRST_N_ * NMS_FC_ * sizeof(_Float16));
  float*    att          = (float*)alloc((size_t)B_ * C_ * FIRST_N_ * NMS_FC_ * sizeof(float));

  // K0: refine boxes
  refine_kernel<<<(B_ * NB_ * C_ + 255) / 256, 256, 0, stream>>>(deltas, pboxes, refined);
  // K1: softmax + top-128 sort + gather boxes (one block per (b,c))
  softmax_sort_kernel<<<B_ * C_, 256, 0, stream>>>(scores, refined, sorted_score,
                                                   rank_idx, sorted_boxes);
  // K2: f16 converts
  f32_to_f16<<<(B_ * NB_ * FEAT_ + 255) / 256, 256, 0, stream>>>(roi_feat, roi_feat_h, B_ * NB_ * FEAT_);
  f32_to_f16<<<(FEAT_ * NMS_FC_ + 255) / 256, 256, 0, stream>>>(roi_w, roi_w_h, FEAT_ * NMS_FC_);
  f32_to_f16<<<(FEAT_ * NMS_FC_ + 255) / 256, 256, 0, stream>>>(rank_w, rank_w_h, FEAT_ * NMS_FC_);
  f32_to_f16<<<(NMS_FC_ * 1024 + 255) / 256, 256, 0, stream>>>(q_w, qw_h, NMS_FC_ * 1024);
  f32_to_f16<<<(NMS_FC_ * 1024 + 255) / 256, 256, 0, stream>>>(k_w, kw_h, NMS_FC_ * 1024);
  // K3: rank embedding
  rank_embed_kernel<<<(FIRST_N_ * FEAT_ + 255) / 256, 256, 0, stream>>>(re_h);
  // K4: WMMA GEMMs
  gemm_f16_wmma<<<dim3((B_ * NB_) / 32, NMS_FC_ / 32), 128, 0, stream>>>(
      roi_feat_h, roi_w_h, roi_b, roi_emb, B_ * NB_, NMS_FC_, FEAT_);
  gemm_f16_wmma<<<dim3(FIRST_N_ / 32, NMS_FC_ / 32), 128, 0, stream>>>(
      re_h, rank_w_h, rank_b, rank_feat, FIRST_N_, NMS_FC_, FEAT_);
  // K5: gather embeddings
  gather_emb_kernel<<<(B_ * C_ * FIRST_N_ * NMS_FC_ + 255) / 256, 256, 0, stream>>>(
      roi_emb, rank_feat, rank_idx, emb_f32, emb_h);
  // K6: fused attention, ~200KB dynamic LDS per workgroup (CDNA5 WGP: 320KB)
  const size_t att_smem = 199680 + 256;
  attention_kernel<<<B_ * C_, 256, att_smem, stream>>>(
      emb_h, sorted_boxes, qw_h, q_b, kw_h, k_b, pos_w, pos_b, out_w, att);
  // K7: final logits
  final_kernel<<<(B_ * C_ * FIRST_N_ + 255) / 256, 256, 0, stream>>>(
      emb_f32, att, out_b, logit_w, logit_b, sorted_score, out);
}
